// Attention_58136677318869
// MI455X (gfx1250) — matrix operand order repaired and checked by reading, not compiled, not run on hardware
//
#include <hip/hip_runtime.h>
#include <cstdint>
#include <cstddef>

#define DIM    2048
#define NHEADS 16
#define HD     128
#define SEQ    2048
#define BATCH  2
#define ROWS   (BATCH * SEQ)   // 4096

typedef __bf16 bf16;
typedef __attribute__((ext_vector_type(16))) __bf16 v16bf;
typedef __attribute__((ext_vector_type(8)))  float  v8f;
typedef int v4i __attribute__((vector_size(16)));

struct __align__(16) U4 { unsigned int x, y, z, w; };
struct U4x2 { U4 lo, hi; };
struct __align__(16) F4 { float x, y, z, w; };

// -------- CDNA5 async global->LDS copy (ASYNCcnt-tracked), with safe fallback
#if defined(__has_builtin)
#  if __has_builtin(__builtin_amdgcn_global_load_async_to_lds_b128)
#    define USE_ASYNC_LDS 1
#  endif
#endif
#ifndef USE_ASYNC_LDS
#  define USE_ASYNC_LDS 0
#endif

__device__ __forceinline__ void async_cp16(void* lds, const void* g) {
#if USE_ASYNC_LDS
  __builtin_amdgcn_global_load_async_to_lds_b128((v4i*)g, (v4i*)lds, 0, 0);
#else
  *(U4*)lds = *(const U4*)g;
#endif
}
__device__ __forceinline__ void wait_async() {
#if USE_ASYNC_LDS
  asm volatile("s_wait_asynccnt 0x0" ::: "memory");
#endif
}

__device__ __forceinline__ bf16 f2bf(float f) {
  unsigned int u = __builtin_bit_cast(unsigned int, f);
  u += 0x7fffu + ((u >> 16) & 1u);              // round-to-nearest-even
  unsigned short h = (unsigned short)(u >> 16);
  return __builtin_bit_cast(bf16, h);
}
__device__ __forceinline__ float bf2f(bf16 b) {
  unsigned short h = __builtin_bit_cast(unsigned short, b);
  unsigned int u = ((unsigned int)h) << 16;
  return __builtin_bit_cast(float, u);
}
// Build a 16 x bf16 WMMA fragment from two 16-byte chunks.
__device__ __forceinline__ v16bf frag16(const bf16* p0, const bf16* p1) {
  U4x2 t;
  t.lo = *(const U4*)p0;
  t.hi = *(const U4*)p1;
  return __builtin_bit_cast(v16bf, t);
}

// ---------------------------------------------------------------- converts
__global__ __launch_bounds__(256)
void convert_f32_to_bf16(const float* __restrict__ in, bf16* __restrict__ out, size_t n) {
  size_t i = ((size_t)blockIdx.x * blockDim.x + threadIdx.x) * 4;
  if (i + 3 < n) {
    F4 v = *(const F4*)(in + i);
    out[i + 0] = f2bf(v.x); out[i + 1] = f2bf(v.y);
    out[i + 2] = f2bf(v.z); out[i + 3] = f2bf(v.w);
  }
}

// out[c*R + r] = bf16(in[r*C + c])  -- one-time weight transpose (LDS tiled)
__global__ __launch_bounds__(256)
void convert_transpose_f32_bf16(const float* __restrict__ in, bf16* __restrict__ out,
                                int R, int C) {
  __shared__ float t[32][33];
  const int c0 = blockIdx.x * 32, r0 = blockIdx.y * 32;
  const int tx = threadIdx.x, ty = threadIdx.y;
#pragma unroll
  for (int i = ty; i < 32; i += 8)
    t[i][tx] = in[(size_t)(r0 + i) * C + c0 + tx];
  __syncthreads();
#pragma unroll
  for (int i = ty; i < 32; i += 8)
    out[(size_t)(c0 + i) * R + r0 + tx] = f2bf(t[tx][i]);
}

// Vt[(b*H+h)*HD + hd][s] = V[b*S + s][h*HD + hd]  -- one-time V transpose
__global__ __launch_bounds__(256)
void transpose_v_bf16(const bf16* __restrict__ V, bf16* __restrict__ Vt) {
  __shared__ bf16 t[32][33];
  const int s0 = blockIdx.x * 32, d0 = blockIdx.y * 32;
  const int bh = blockIdx.z;
  const int b = bh >> 4, h = bh & 15;
  const int tx = threadIdx.x, ty = threadIdx.y;
#pragma unroll
  for (int i = ty; i < 32; i += 8)
    t[i][tx] = V[((size_t)b * SEQ + s0 + i) * DIM + (size_t)h * HD + d0 + tx];
  __syncthreads();
#pragma unroll
  for (int i = ty; i < 32; i += 8)
    Vt[((size_t)bh * HD + d0 + i) * SEQ + s0 + tx] = t[tx][i];
}

// ---------------------------------------------------------------- GEMM
__device__ __forceinline__ void store_out(float* p, float v) { *p = v; }
__device__ __forceinline__ void store_out(bf16*  p, float v) { *p = f2bf(v); }

// C[M,N] = A[M,K] (row-major bf16) * Bt[N,K] (PRE-TRANSPOSED bf16), f32 accumulate.
// Block tile 128x256, 8 waves as 2x4, wave tile 64x64 (4x4 WMMAs per K=32 step).
// Double-buffered LDS; staging uses async global->LDS copies.
template <typename OutT>
__global__ __launch_bounds__(256)
void gemm_bf16_wmma(const bf16* __restrict__ A, const bf16* __restrict__ Bt,
                    OutT* __restrict__ C, int M, int N, int K) {
  __shared__ bf16 As[2][128][32];   // 2 x 8 KB
  __shared__ bf16 Bs[2][256][32];   // 2 x 16 KB
  const int tid  = threadIdx.x;
  const int lane = tid & 31;
  const int wave = tid >> 5;
  const int wm   = wave >> 2;       // 0..1 -> 64-row slab
  const int wn   = wave & 3;        // 0..3 -> 64-col slab
  const int ll   = lane & 15;
  const int lg   = lane >> 4;
  const int m0   = blockIdx.y * 128;
  const int n0   = blockIdx.x * 256;

  v8f acc[4][4] = {};

  const int ar = tid >> 1;          // 0..127
  const int ac = (tid & 1) * 16;    // 0 or 16
  const int br = tid;               // 0..255

  auto stage = [&](int buf, int kb) {
    const bf16* sa = A + (size_t)(m0 + ar) * K + kb + ac;
    async_cp16(&As[buf][ar][ac],     sa);
    async_cp16(&As[buf][ar][ac + 8], sa + 8);
    const bf16* sb = Bt + (size_t)(n0 + br) * K + kb;
    async_cp16(&Bs[buf][br][0],  sb);
    async_cp16(&Bs[buf][br][8],  sb + 8);
    async_cp16(&Bs[buf][br][16], sb + 16);
    async_cp16(&Bs[buf][br][24], sb + 24);
  };

  stage(0, 0);
  for (int kb = 0; kb < K; kb += 32) {
    const int cur = (kb >> 5) & 1;
    wait_async();                             // staged copies landed in LDS
    __syncthreads();                          // visible to all waves; prev compute done
    if (kb + 32 < K) stage(cur ^ 1, kb + 32); // prefetch next tile into other buffer

    v16bf af[4], bfr[4];
#pragma unroll
    for (int mi = 0; mi < 4; ++mi) {
      const bf16* p = &As[cur][wm * 64 + mi * 16 + ll][lg * 8];   // K: lg*8..+7, +16..+23
      af[mi] = frag16(p, p + 16);
    }
#pragma unroll
    for (int ni = 0; ni < 4; ++ni) {
      const bf16* p = &Bs[cur][wn * 64 + ni * 16 + ll][lg * 8];   // K: lg*8..+7, +16..+23
      bfr[ni] = frag16(p, p + 16);
    }
#pragma unroll
    for (int mi = 0; mi < 4; ++mi)
#pragma unroll
      for (int ni = 0; ni < 4; ++ni)
        acc[mi][ni] = __builtin_amdgcn_wmma_f32_16x16x32_bf16(
            false, af[mi], false, bfr[ni], (short)0, acc[mi][ni], false, false);
  }

#pragma unroll
  for (int mi = 0; mi < 4; ++mi)
#pragma unroll
    for (int ni = 0; ni < 4; ++ni)
#pragma unroll
      for (int r = 0; r < 8; ++r) {
        int row = m0 + wm * 64 + mi * 16 + r + lg * 8;
        int col = n0 + wn * 64 + ni * 16 + ll;
        store_out(&C[(size_t)row * N + col], acc[mi][ni][r]);
      }
}

// ---------------------------------------------------------------- RoPE (in-place on bf16 Q,K)
__global__ __launch_bounds__(256)
void rope_inplace(bf16* __restrict__ q, bf16* __restrict__ k,
                  const float* __restrict__ cost, const float* __restrict__ sint) {
  size_t idx = (size_t)blockIdx.x * blockDim.x + threadIdx.x;  // B*S*H*64 threads
  int    i   = (int)(idx & 63);
  size_t t   = idx >> 6;
  int    h   = (int)(t & (NHEADS - 1));
  size_t bs  = t >> 4;
  int    s   = (int)(bs & (SEQ - 1));
  size_t base = bs * (size_t)DIM + (size_t)h * HD + 2 * i;
  float c  = cost[(size_t)s * 64 + i];
  float sn = sint[(size_t)s * 64 + i];
  float q0 = bf2f(q[base]), q1 = bf2f(q[base + 1]);
  q[base]     = f2bf(q0 * c - q1 * sn);
  q[base + 1] = f2bf(q0 * sn + q1 * c);
  float k0 = bf2f(k[base]), k1 = bf2f(k[base + 1]);
  k[base]     = f2bf(k0 * c - k1 * sn);
  k[base + 1] = f2bf(k0 * sn + k1 * c);
}

// ---------------------------------------------------------------- Flash attention
// Block = (b, h, 128-query slab); 8 waves, each owns 16 query rows + private online softmax.
// Vt_full is the pre-transposed V: [(b*H+h)*HD + hd][SEQ].
// K/V^T chunks are double-buffered with async global->LDS staging.
__global__ __launch_bounds__(256)
void flash_attn_wmma(const bf16* __restrict__ Q, const bf16* __restrict__ K,
                     const bf16* __restrict__ Vt_full, bf16* __restrict__ O) {
  __shared__ bf16 Ks[2][32][128];  // kv chunk, row-major [key][hd]    2 x 8 KB
  __shared__ bf16 Vt[2][128][32];  // V^T chunk [hd][key]              2 x 8 KB
  __shared__ bf16 Ps[8][16][32];   // per-wave P bounce buffer (C-layout -> A-layout)

  const int tid  = threadIdx.x;
  const int lane = tid & 31;
  const int wave = tid >> 5;
  const int ll   = lane & 15;
  const int lg   = lane >> 4;

  const int qblk = blockIdx.x;     // 0..15
  const int h    = blockIdx.y;
  const int b    = blockIdx.z;

  const int    qbase   = qblk * 128 + wave * 16;
  const size_t rowbase = (size_t)b * SEQ;
  const bf16*  vtbase  = Vt_full + (size_t)(b * NHEADS + h) * HD * SEQ;

  // Q fragments: 16 rows x 128 hd as 4 K=32 chunks
  v16bf qf[4];
  {
    const bf16* qp = Q + (rowbase + qbase + ll) * (size_t)DIM + (size_t)h * HD;
#pragma unroll
    for (int c = 0; c < 4; ++c) {
      const bf16* p = qp + c * 32 + lg * 8;
      qf[c] = frag16(p, p + 16);
    }
  }

  v8f o[8] = {};
  float mrow[8], lrow[8];
#pragma unroll
  for (int r = 0; r < 8; ++r) { mrow[r] = -3.0e38f; lrow[r] = 0.0f; }

  const float scale  = 0.08838834764831845f;   // 1/sqrt(128)
  const int   kv_end = qblk * 128 + 128;       // causal: uniform per block
  const int   kr = tid >> 3;                   // 0..31
  const int   kc = (tid & 7) * 16;             // 0..112
  const int   vr = tid >> 1;                   // 0..127
  const int   vc = (tid & 1) * 16;             // 0 or 16

  auto stageKV = [&](int buf, int kv0) {
    const bf16* ksrc = K + (rowbase + kv0 + kr) * (size_t)DIM + (size_t)h * HD + kc;
    async_cp16(&Ks[buf][kr][kc],     ksrc);
    async_cp16(&Ks[buf][kr][kc + 8], ksrc + 8);
    const bf16* vsrc = vtbase + (size_t)vr * SEQ + kv0 + vc;
    async_cp16(&Vt[buf][vr][vc],     vsrc);
    async_cp16(&Vt[buf][vr][vc + 8], vsrc + 8);
  };

  stageKV(0, 0);
  for (int kv0 = 0; kv0 < kv_end; kv0 += 32) {
    const int cur = (kv0 >> 5) & 1;
    wait_async();                                    // staged K/V^T landed
    __syncthreads();                                 // visible; prev chunk's reads done
    if (kv0 + 32 < kv_end) stageKV(cur ^ 1, kv0 + 32);

    // S = Q K^T : two 16x16 key tiles, K-dim = hd accumulated over 4 WMMAs
    v8f sc[2] = {};
#pragma unroll
    for (int t2 = 0; t2 < 2; ++t2) {
#pragma unroll
      for (int c = 0; c < 4; ++c) {
        const bf16* p = &Ks[cur][t2 * 16 + ll][c * 32 + lg * 8];
        v16bf kf = frag16(p, p + 16);
        sc[t2] = __builtin_amdgcn_wmma_f32_16x16x32_bf16(
            false, qf[c], false, kf, (short)0, sc[t2], false, false);
      }
    }

    // online softmax (row-wise; rows live in lane halves per C layout)
#pragma unroll
    for (int r = 0; r < 8; ++r) {
      const int row = qbase + r + lg * 8;
      float s0 = sc[0][r] * scale;
      float s1 = sc[1][r] * scale;
      if (kv0 + ll > row)      s0 = -3.0e38f;   // causal mask
      if (kv0 + 16 + ll > row) s1 = -3.0e38f;
      float mx = fmaxf(s0, s1);
#pragma unroll
      for (int msk = 1; msk < 16; msk <<= 1) mx = fmaxf(mx, __shfl_xor(mx, msk, 32));
      float mnew  = fmaxf(mrow[r], mx);
      float alpha = __expf(mrow[r] - mnew);
      mrow[r] = mnew;
      float p0 = __expf(s0 - mnew);
      float p1 = __expf(s1 - mnew);
      float ps = p0 + p1;
#pragma unroll
      for (int msk = 1; msk < 16; msk <<= 1) ps += __shfl_xor(ps, msk, 32);
      lrow[r] = lrow[r] * alpha + ps;
#pragma unroll
      for (int ni = 0; ni < 8; ++ni) o[ni][r] *= alpha;
      Ps[wave][r + lg * 8][ll]      = f2bf(p0);
      Ps[wave][r + lg * 8][16 + ll] = f2bf(p1);
    }
    __syncthreads();

    // O += P @ V   (k = 32 keys)
    {
      const bf16* pp = &Ps[wave][ll][lg * 8];
      v16bf pf = frag16(pp, pp + 16);
#pragma unroll
      for (int ni = 0; ni < 8; ++ni) {
        const bf16* vp = &Vt[cur][ni * 16 + ll][lg * 8];
        v16bf vf = frag16(vp, vp + 16);
        o[ni] = __builtin_amdgcn_wmma_f32_16x16x32_bf16(
            false, pf, false, vf, (short)0, o[ni], false, false);
      }
    }
  }

  // normalize & write O (bf16, feeds the wo projection GEMM)
#pragma unroll
  for (int r = 0; r < 8; ++r) {
    float  inv = 1.0f / lrow[r];
    size_t row = rowbase + qbase + r + lg * 8;
#pragma unroll
    for (int ni = 0; ni < 8; ++ni)
      O[row * (size_t)DIM + (size_t)h * HD + ni * 16 + ll] = f2bf(o[ni][r] * inv);
  }
}

// ---------------------------------------------------------------- launch
extern "C" void kernel_launch(void* const* d_in, const int* in_sizes, int n_in,
                              void* d_out, int out_size, void* d_ws, size_t ws_size,
                              hipStream_t stream) {
  (void)in_sizes; (void)n_in; (void)out_size; (void)ws_size;
  const float* x  = (const float*)d_in[0];
  const float* wq = (const float*)d_in[1];
  const float* wk = (const float*)d_in[2];
  const float* wv = (const float*)d_in[3];
  const float* wo = (const float*)d_in[4];
  const float* fc = (const float*)d_in[5];
  const float* fs = (const float*)d_in[6];

  const size_t nx = (size_t)ROWS * DIM;   // 8,388,608
  const size_t nw = (size_t)DIM * DIM;    // 4,194,304

  char* ws = (char*)d_ws;
  size_t off = 0;
  bf16* xbf  = (bf16*)(ws + off); off += nx * 2;
  bf16* wqT  = (bf16*)(ws + off); off += nw * 2;   // pre-transposed [N][K]
  bf16* wkT  = (bf16*)(ws + off); off += nw * 2;
  bf16* wvT  = (bf16*)(ws + off); off += nw * 2;
  bf16* woT  = (bf16*)(ws + off); off += nw * 2;
  bf16* Qb   = (bf16*)(ws + off); off += nx * 2;
  bf16* Kb   = (bf16*)(ws + off); off += nx * 2;
  bf16* Vb   = (bf16*)(ws + off); off += nx * 2;
  bf16* VbT  = (bf16*)(ws + off); off += nx * 2;   // [(b,h)][HD][SEQ]
  bf16* Ob   = (bf16*)(ws + off); off += nx * 2;   // total ~128 MB

  dim3 tblk(32, 8);
  convert_f32_to_bf16<<<(unsigned)(nx / 1024), 256, 0, stream>>>(x, xbf, nx);
  convert_transpose_f32_bf16<<<dim3(64, 64), tblk, 0, stream>>>(wq, wqT, DIM, DIM);
  convert_transpose_f32_bf16<<<dim3(64, 64), tblk, 0, stream>>>(wk, wkT, DIM, DIM);
  convert_transpose_f32_bf16<<<dim3(64, 64), tblk, 0, stream>>>(wv, wvT, DIM, DIM);
  convert_transpose_f32_bf16<<<dim3(64, 64), tblk, 0, stream>>>(wo, woT, DIM, DIM);

  dim3 gg(DIM / 256, ROWS / 128);          // (8, 32)
  gemm_bf16_wmma<bf16><<<gg, 256, 0, stream>>>(xbf, wqT, Qb, ROWS, DIM, DIM);
  gemm_bf16_wmma<bf16><<<gg, 256, 0, stream>>>(xbf, wkT, Kb, ROWS, DIM, DIM);
  gemm_bf16_wmma<bf16><<<gg, 256, 0, stream>>>(xbf, wvT, Vb, ROWS, DIM, DIM);

  rope_inplace<<<(ROWS * NHEADS * 64) / 256, 256, 0, stream>>>(Qb, Kb, fc, fs);
  transpose_v_bf16<<<dim3(SEQ / 32, HD / 32, BATCH * NHEADS), tblk, 0, stream>>>(Vb, VbT);

  flash_attn_wmma<<<dim3(SEQ / 128, NHEADS, BATCH), 256, 0, stream>>>(Qb, Kb, VbT, Ob);

  gemm_bf16_wmma<float><<<gg, 256, 0, stream>>>(Ob, woT, (float*)d_out, ROWS, DIM, DIM);
}
